// GNN_49297634624011
// MI455X (gfx1250) — compile-verified, matrix operand order checked by reading, not run on hardware
//
#include <hip/hip_runtime.h>
#include <hip/hip_bf16.h>

// ---------------------------------------------------------------------------
// GIN (pretrain-gnns) 5-layer forward for MI455X (gfx1250, wave32, WMMA).
// GEMMs: v_wmma_f32_16x16x32_bf16, 128x128 block tiles, 8 waves x (32x64).
// A tiles are DMA'd by the Tensor Data Mover (tensor_load_to_lds with D#
// LDS-padding -> KPAD row stride), double-buffered against the WMMA loop and
// synchronized with s_wait_tensorcnt. B tiles are cooperatively converted
// f32->bf16 with a branch-free fast path for interior tiles.
// ---------------------------------------------------------------------------

#define NNODES 150000
#define NEDGES 300000
#define DIM    300     // D
#define HID    600     // 2D
#define NLAYER 5
#define BN_EPS 1e-5f

typedef __attribute__((ext_vector_type(16))) __bf16 v16bf;
typedef __attribute__((ext_vector_type(8)))  __bf16 v8bf;
typedef __attribute__((ext_vector_type(4)))  __bf16 v4bf;
typedef __attribute__((ext_vector_type(8)))  float  v8f;
typedef __attribute__((ext_vector_type(4)))  unsigned int u32x4;
typedef __attribute__((ext_vector_type(8)))  int i32x8;
typedef __attribute__((ext_vector_type(4)))  int i32x4;

#define BM   128   // M tile per workgroup
#define BNT  128   // N tile per workgroup
#define BK   32    // K step (one bf16 WMMA K)
#define KPAD 40    // LDS row stride in bf16 (80B = 20 DWORDs: 16 data + 4 pad)

#if defined(__has_builtin)
#if __has_builtin(__builtin_amdgcn_tensor_load_to_lds) && \
    __has_builtin(__builtin_amdgcn_s_wait_tensorcnt)
#define HAVE_TDM 1
#endif
#endif

// ---------------------------------------------------------------------------
// Fragment loader (ISA 7.12.2, 16-bit A 16x32 layout):
//   lanes 0-15  : row M=lane,    elems {K0..K7,  K16..K23}
//   lanes 16-31 : row M=lane-16, elems {K8..K15, K24..K31}
// ---------------------------------------------------------------------------
static __device__ __forceinline__ v16bf load_frag(const __bf16* rowbase, int lane) {
  int row = lane & 15;
  int kb  = (lane >> 4) << 3;            // 0 or 8
  const __bf16* p = rowbase + row * KPAD + kb;
  v8bf lo = *(const v8bf*)(p);           // K kb..kb+7
  v8bf hi = *(const v8bf*)(p + 16);      // K kb+16..kb+23
  v16bf r;
#pragma unroll
  for (int i = 0; i < 8; ++i) { r[i] = lo[i]; r[i + 8] = hi[i]; }
  return r;
}

static __device__ __forceinline__ v8f wmma_bf16(v16bf a, v16bf b, v8f c) {
  return __builtin_amdgcn_wmma_f32_16x16x32_bf16(false, a, false, b, (short)0, c,
                                                 false, false);
}

#ifdef HAVE_TDM
// ---------------------------------------------------------------------------
// TDM: DMA a [rows x 32] bf16 tile (memory row stride = lda elems) into LDS,
// padding 4 DWORDs after every 16 DWORDs -> LDS row stride = KPAD bf16.
// D# per CDNA5 ISA ch.8. Issued by one wave (EXEC ignored by tensor ops).
// ---------------------------------------------------------------------------
static __device__ __forceinline__ void tdm_load_a_tile(const __bf16* tile_start,
                                                       unsigned lds_byte,
                                                       int rows, int lda) {
  unsigned long long ga = (unsigned long long)(uintptr_t)(const void*)tile_start;
  u32x4 g0;
  g0[0] = 1u;                                             // count=1, user desc
  g0[1] = lds_byte;                                       // lds_addr (bytes)
  g0[2] = (unsigned)(ga & 0xFFFFFFFFull);                 // global_addr[31:0]
  g0[3] = (unsigned)((ga >> 32) & 0x01FFFFFFull) | (2u << 30);  // [56:32]|type=2
  i32x8 g1;
  // data_size=1(2B) | pad_enable | pad_interval=3(16 DW) | pad_amount=3(4 DW)
  g1[0] = (1 << 16) | (1 << 20) | (3 << 22) | (3 << 25);
  g1[1] = (int)(((unsigned)lda & 0xFFFFu) << 16);         // tensor_dim0 lo16
  g1[2] = (int)(((unsigned)lda >> 16) |
                (((unsigned)NNODES & 0xFFFFu) << 16));    // dim0 hi | dim1 lo16
  g1[3] = (int)(((unsigned)NNODES >> 16) | (32u << 16));  // dim1 hi | tile_dim0=32
  g1[4] = rows;                                           // tile_dim1 | tile_dim2=0
  g1[5] = lda;                                            // tensor_dim0_stride lo32
  g1[6] = 0;
  g1[7] = 0;
  i32x4 gz;
  gz[0] = 0; gz[1] = 0; gz[2] = 0; gz[3] = 0;
#if __clang_major__ >= 23
  i32x8 gz8;
#pragma unroll
  for (int i = 0; i < 8; ++i) gz8[i] = 0;
  __builtin_amdgcn_tensor_load_to_lds(g0, g1, gz, gz, gz8, 0);
#else
  __builtin_amdgcn_tensor_load_to_lds(g0, g1, gz, gz, 0);
#endif
}
#endif // HAVE_TDM

// ---------------------------------------------------------------------------
// 1. Embedding sum: h[n,:] = x_emb[id0,:] + x_emb[id1,:]
// ---------------------------------------------------------------------------
__global__ void embed_kernel(const int* __restrict__ x_ids,
                             const float* __restrict__ x_emb,
                             float* __restrict__ h) {
  int idx = blockIdx.x * blockDim.x + threadIdx.x;       // over N * 75 float4s
  if (idx >= NNODES * (DIM / 4)) return;
  int n  = idx / (DIM / 4);
  int c4 = (idx - n * (DIM / 4)) * 4;
  int id0 = x_ids[n * 2 + 0];
  int id1 = x_ids[n * 2 + 1];
  float4 a = *(const float4*)(x_emb + (size_t)id0 * DIM + c4);
  float4 b = *(const float4*)(x_emb + (size_t)id1 * DIM + c4);
  float4 o = make_float4(a.x + b.x, a.y + b.y, a.z + b.z, a.w + b.w);
  *(float4*)(h + (size_t)n * DIM + c4) = o;
}

// ---------------------------------------------------------------------------
// 2. agg = h + e1[l,4,:] + e2[l,0,:]   (self-loop contribution + identity)
// ---------------------------------------------------------------------------
__global__ void init_agg_kernel(const float* __restrict__ h,
                                const float* __restrict__ e1_self,
                                const float* __restrict__ e2_self,
                                float* __restrict__ agg) {
  int idx = blockIdx.x * blockDim.x + threadIdx.x;
  if (idx >= NNODES * (DIM / 4)) return;
  int n  = idx / (DIM / 4);
  int c4 = (idx - n * (DIM / 4)) * 4;
  float4 hv = *(const float4*)(h + (size_t)n * DIM + c4);
  float4 s1 = *(const float4*)(e1_self + c4);
  float4 s2 = *(const float4*)(e2_self + c4);
  float4 o = make_float4(hv.x + s1.x + s2.x, hv.y + s1.y + s2.y,
                         hv.z + s1.z + s2.z, hv.w + s1.w + s2.w);
  *(float4*)(agg + (size_t)n * DIM + c4) = o;
}

// ---------------------------------------------------------------------------
// 3. Edge scatter-add: agg[dst] += h[src] + e1[l,ea0] + e2[l,ea1]
// ---------------------------------------------------------------------------
__global__ void scatter_kernel(const int* __restrict__ ei,
                               const int* __restrict__ ea,
                               const float* __restrict__ h,
                               const float* __restrict__ e1l,
                               const float* __restrict__ e2l,
                               float* __restrict__ agg) {
  int idx = blockIdx.x * blockDim.x + threadIdx.x;       // over E * 75 float4s
  if (idx >= NEDGES * (DIM / 4)) return;
  int e  = idx / (DIM / 4);
  int c4 = (idx - e * (DIM / 4)) * 4;
  int src = ei[e];
  int dst = ei[NEDGES + e];
  int a0  = ea[e * 2 + 0];
  int a1  = ea[e * 2 + 1];
  float4 hv = *(const float4*)(h + (size_t)src * DIM + c4);
  float4 v1 = *(const float4*)(e1l + (size_t)a0 * DIM + c4);
  float4 v2 = *(const float4*)(e2l + (size_t)a1 * DIM + c4);
  float* p = agg + (size_t)dst * DIM + c4;
  atomicAdd(p + 0, hv.x + v1.x + v2.x);
  atomicAdd(p + 1, hv.y + v1.y + v2.y);
  atomicAdd(p + 2, hv.z + v1.z + v2.z);
  atomicAdd(p + 3, hv.w + v1.w + v2.w);
}

// ---------------------------------------------------------------------------
// 3b. agg(f32) -> aggbf(bf16) once per layer (GEMM1 A operand; TDM-friendly)
// ---------------------------------------------------------------------------
__global__ void convert_bf16_kernel(const float* __restrict__ src,
                                    __bf16* __restrict__ dst) {
  int idx = blockIdx.x * blockDim.x + threadIdx.x;
  if (idx >= NNODES * (DIM / 4)) return;
  int n  = idx / (DIM / 4);
  int c4 = (idx - n * (DIM / 4)) * 4;
  float4 v = *(const float4*)(src + (size_t)n * DIM + c4);
  v4bf o;
  o[0] = (__bf16)v.x; o[1] = (__bf16)v.y; o[2] = (__bf16)v.z; o[3] = (__bf16)v.w;
  *(v4bf*)(dst + (size_t)n * DIM + c4) = o;
}

// ---------------------------------------------------------------------------
// 4/5. Unified GEMM: out = A[N,K](bf16) @ W[K,NCOLS](f32->bf16) + bias
//   RELU_BF16 ? out=bf16 with ReLU (GEMM1) : out=f32 (GEMM2)
//   Double-buffered LDS; A via TDM, B via cooperative branch-free loads.
// ---------------------------------------------------------------------------
template <int K, int NCOLS, bool RELU_BF16>
__global__ __launch_bounds__(256) void gemm_kernel(const __bf16* __restrict__ A,
                                                   const float* __restrict__ W,
                                                   const float* __restrict__ bias,
                                                   __bf16* __restrict__ outb,
                                                   float* __restrict__ outf) {
  __shared__ __bf16 Alds[2][BM * KPAD];
  __shared__ __bf16 Blds[2][BNT * KPAD];
  const int tid  = threadIdx.x;
  const int lane = tid & 31;
  const int wave = tid >> 5;
  const int wm = wave & 3;          // M offset wm*32
  const int wn = wave >> 2;         // N offset wn*64
  const int mBase = blockIdx.y * BM;
  const int nBase = blockIdx.x * BNT;
  int rows = NNODES - mBase; if (rows > BM) rows = BM;
  (void)rows;

  constexpr int KSTEPS = (K + BK - 1) / BK;

  v8f c[2][4] = {};

  // ---- A tile stage: TDM DMA (or cooperative bf16 copy fallback)
  auto issueA = [&](int kb, int b) {
#ifdef HAVE_TDM
    if (tid < 32)
      tdm_load_a_tile(A + (size_t)mBase * K + kb,
                      (unsigned)(uintptr_t)(void*)&Alds[b][0], rows, K);
#else
#pragma unroll
    for (int it = 0; it < 4; ++it) {
      int slot = tid + it * 256;           // 128 rows x 8 chunks of 4
      int row  = slot >> 3;
      int c4   = (slot & 7) * 4;
      int gr   = mBase + row;
      v4bf av;
      av[0] = av[1] = av[2] = av[3] = (__bf16)0.0f;
      if (gr < NNODES)
        av = *(const v4bf*)(A + (size_t)gr * K + kb + c4);
      *(v4bf*)&Alds[b][row * KPAD + c4] = av;
    }
#endif
  };

  // ---- B tile stage: 32x128 f32 -> bf16, transposed into Blds[n][k]
  auto loadB = [&](int kb, int b) {
    bool full = (kb + BK <= K) && (nBase + BNT <= NCOLS);
    if (full) {                            // branch-free interior path
#pragma unroll
      for (int it = 0; it < 4; ++it) {
        int slot = tid + it * 256;         // 0..1023 : 32 k x 32 n-quads
        int kl = slot >> 5;                // 0..31
        int n4 = (slot & 31) * 4;          // 0..124
        float4 v = *(const float4*)(W + (size_t)(kb + kl) * NCOLS + nBase + n4);
        __bf16* d = &Blds[b][0];
        d[(n4 + 0) * KPAD + kl] = (__bf16)v.x;
        d[(n4 + 1) * KPAD + kl] = (__bf16)v.y;
        d[(n4 + 2) * KPAD + kl] = (__bf16)v.z;
        d[(n4 + 3) * KPAD + kl] = (__bf16)v.w;
      }
    } else {                               // guarded edge path
#pragma unroll
      for (int it = 0; it < 16; ++it) {
        int slot = tid + it * 256;
        int kl = slot >> 7, nl = slot & 127;
        int k = kb + kl, n = nBase + nl;
        float v = 0.f;
        if (k < K && n < NCOLS) v = W[(size_t)k * NCOLS + n];
        Blds[b][nl * KPAD + kl] = (__bf16)v;
      }
    }
  };

  // ---- prologue
  issueA(0, 0);
  loadB(0, 0);
#ifdef HAVE_TDM
  if (tid < 32) __builtin_amdgcn_s_wait_tensorcnt(0);
#endif
  __syncthreads();

  // ---- pipelined main loop
  int buf = 0;
  for (int s = 0; s < KSTEPS; ++s) {
    if (s + 1 < KSTEPS) {
      issueA((s + 1) * BK, buf ^ 1);       // DMA overlaps the WMMAs below
      loadB((s + 1) * BK, buf ^ 1);
      if ((s + 2) * BK < K)                // prefetch the K-slab after next
        __builtin_prefetch(W + (size_t)(s + 2) * BK * NCOLS + nBase + (tid & 127),
                           0, 1);
    }
    v16bf a0 = load_frag(&Alds[buf][(wm * 32 + 0)  * KPAD], lane);
    v16bf a1 = load_frag(&Alds[buf][(wm * 32 + 16) * KPAD], lane);
#pragma unroll
    for (int j = 0; j < 4; ++j) {
      v16bf bf = load_frag(&Blds[buf][(wn * 64 + j * 16) * KPAD], lane);
      c[0][j] = wmma_bf16(a0, bf, c[0][j]);
      c[1][j] = wmma_bf16(a1, bf, c[1][j]);
    }
#ifdef HAVE_TDM
    if (s + 1 < KSTEPS && tid < 32) __builtin_amdgcn_s_wait_tensorcnt(0);
#endif
    __syncthreads();
    buf ^= 1;
  }

  // ---- epilogue: C layout: VGPR r -> lanes0-15 M=r, lanes16-31 M=8+r; N=lane&15
  const int mw = mBase + wm * 32;
  const int nw = nBase + wn * 64;
#pragma unroll
  for (int i = 0; i < 2; ++i) {
#pragma unroll
    for (int j = 0; j < 4; ++j) {
      int n = nw + j * 16 + (lane & 15);
      float bv = (n < NCOLS) ? bias[n] : 0.f;
      int mrow = mw + i * 16 + ((lane >> 4) * 8);
#pragma unroll
      for (int r = 0; r < 8; ++r) {
        int m = mrow + r;
        if (m < NNODES && n < NCOLS) {
          float v = c[i][j][r] + bv;
          if constexpr (RELU_BF16)
            outb[(size_t)m * NCOLS + n] = (__bf16)fmaxf(v, 0.f);
          else
            outf[(size_t)m * NCOLS + n] = v;
        }
      }
    }
  }
}

// ---------------------------------------------------------------------------
// 6-9. BatchNorm (training-mode batch stats) + ELU
// stats layout: [0..319]=sum, [320..639]=sumsq, [640..959]=scale, [960..1279]=shift
// ---------------------------------------------------------------------------
__global__ void zero_stats_kernel(float* __restrict__ stats) {
  int t = threadIdx.x;
  if (t < 640) stats[t] = 0.f;
}

#define RED_CHUNK 512
__global__ void bn_reduce_kernel(const float* __restrict__ h2,
                                 float* __restrict__ stats) {
  int d = threadIdx.x;
  if (d >= DIM) return;
  int r0 = blockIdx.x * RED_CHUNK;
  int r1 = r0 + RED_CHUNK; if (r1 > NNODES) r1 = NNODES;
  float s = 0.f, q = 0.f;
  for (int r = r0; r < r1; ++r) {
    float v = h2[(size_t)r * DIM + d];     // consecutive d -> coalesced
    s += v; q += v * v;
  }
  atomicAdd(&stats[d], s);
  atomicAdd(&stats[320 + d], q);
}

__global__ void bn_finalize_kernel(const float* __restrict__ gamma,
                                   const float* __restrict__ beta,
                                   float* __restrict__ stats) {
  int d = threadIdx.x;
  if (d >= DIM) return;
  const float invN = 1.0f / (float)NNODES;
  float mean = stats[d] * invN;
  float var  = stats[320 + d] * invN - mean * mean;    // biased, matches ref
  float inv  = rsqrtf(var + BN_EPS);
  float sc   = gamma[d] * inv;
  stats[640 + d] = sc;
  stats[960 + d] = beta[d] - mean * sc;
}

__global__ void bn_apply_kernel(const float* __restrict__ h2,
                                const float* __restrict__ stats,
                                float* __restrict__ out, int apply_elu) {
  int idx = blockIdx.x * blockDim.x + threadIdx.x;
  if (idx >= NNODES * (DIM / 4)) return;
  int n  = idx / (DIM / 4);
  int c4 = (idx - n * (DIM / 4)) * 4;
  float4 v = *(const float4*)(h2 + (size_t)n * DIM + c4);
  float o[4] = {v.x, v.y, v.z, v.w};
#pragma unroll
  for (int u = 0; u < 4; ++u) {
    int d = c4 + u;
    float y = o[u] * stats[640 + d] + stats[960 + d];
    if (apply_elu && y < 0.f) y = __expf(y) - 1.0f;
    o[u] = y;
  }
  *(float4*)(out + (size_t)n * DIM + c4) = make_float4(o[0], o[1], o[2], o[3]);
}

// ---------------------------------------------------------------------------
// Orchestration
// ---------------------------------------------------------------------------
extern "C" void kernel_launch(void* const* d_in, const int* in_sizes, int n_in,
                              void* d_out, int out_size, void* d_ws, size_t ws_size,
                              hipStream_t stream) {
  const int*   x_ids = (const int*)d_in[0];
  const int*   ei    = (const int*)d_in[1];
  const int*   ea    = (const int*)d_in[2];
  const float* x_emb = (const float*)d_in[3];
  const float* e1    = (const float*)d_in[4];   // [L,6,D]
  const float* e2    = (const float*)d_in[5];   // [L,3,D]
  const float* W1    = (const float*)d_in[6];   // [L,D,2D]
  const float* b1    = (const float*)d_in[7];   // [L,2D]
  const float* W2    = (const float*)d_in[8];   // [L,2D,D]
  const float* b2    = (const float*)d_in[9];   // [L,D]
  const float* gamma = (const float*)d_in[10];  // [L,D]
  const float* beta  = (const float*)d_in[11];  // [L,D]

  char* ws = (char*)d_ws;
  const size_t hBytes    = (size_t)NNODES * DIM * sizeof(float);   // 180 MB
  const size_t hidBytes  = (size_t)NNODES * HID * sizeof(__bf16);  // 180 MB
  const size_t abfBytes  = (size_t)NNODES * DIM * sizeof(__bf16);  //  90 MB
  float*  h      = (float*)(ws);
  float*  agg    = (float*)(ws + hBytes);            // also aliases h2
  __bf16* hidden = (__bf16*)(ws + 2 * hBytes);       // K-edge overrun -> aggbf
  __bf16* aggbf  = (__bf16*)(ws + 2 * hBytes + hidBytes); // overrun -> stats
  float*  stats  = (float*)(ws + 2 * hBytes + hidBytes + abfBytes);

  const int TPB = 256;
  const int nElem4 = NNODES * (DIM / 4);             // 11.25M
  const int eElem4 = NEDGES * (DIM / 4);             // 22.5M
  const int gN = (nElem4 + TPB - 1) / TPB;
  const int gE = (eElem4 + TPB - 1) / TPB;
  const dim3 grid1((HID + BNT - 1) / BNT, (NNODES + BM - 1) / BM);  // (5, 1172)
  const dim3 grid2((DIM + BNT - 1) / BNT, (NNODES + BM - 1) / BM);  // (3, 1172)
  const int gRed = (NNODES + RED_CHUNK - 1) / RED_CHUNK;

  embed_kernel<<<gN, TPB, 0, stream>>>(x_ids, x_emb, h);

  for (int l = 0; l < NLAYER; ++l) {
    const float* e1l = e1 + (size_t)l * 6 * DIM;
    const float* e2l = e2 + (size_t)l * 3 * DIM;
    const float* e1_self = e1l + (size_t)4 * DIM;    // SELF_LOOP_BOND_TYPE = 4
    const float* e2_self = e2l;                      // SELF_LOOP_BOND_DIR  = 0

    init_agg_kernel<<<gN, TPB, 0, stream>>>(h, e1_self, e2_self, agg);
    scatter_kernel<<<gE, TPB, 0, stream>>>(ei, ea, h, e1l, e2l, agg);
    convert_bf16_kernel<<<gN, TPB, 0, stream>>>(agg, aggbf);

    gemm_kernel<DIM, HID, true><<<grid1, TPB, 0, stream>>>(
        aggbf, W1 + (size_t)l * DIM * HID, b1 + (size_t)l * HID, hidden, nullptr);

    gemm_kernel<HID, DIM, false><<<grid2, TPB, 0, stream>>>(
        hidden, W2 + (size_t)l * HID * DIM, b2 + (size_t)l * DIM, nullptr,
        agg /*h2*/);

    zero_stats_kernel<<<1, 640, 0, stream>>>(stats);
    bn_reduce_kernel<<<gRed, 320, 0, stream>>>(agg, stats);
    bn_finalize_kernel<<<1, 320, 0, stream>>>(
        gamma + (size_t)l * DIM, beta + (size_t)l * DIM, stats);

    float* dst = (l == NLAYER - 1) ? (float*)d_out : h;
    bn_apply_kernel<<<gN, TPB, 0, stream>>>(agg, stats, dst, (l < NLAYER - 1) ? 1 : 0);
  }
}
